// MaskedMultiQuerySelfAttention_34754875359339
// MI455X (gfx1250) — matrix operand order repaired and checked by reading, not compiled, not run on hardware
//
#include <hip/hip_runtime.h>
#include <hip/hip_bf16.h>
#include <stdint.h>

// ---------------------------------------------------------------------------
// Types / helpers
// ---------------------------------------------------------------------------
typedef __bf16 v16bf __attribute__((ext_vector_type(16)));
typedef float  v8f   __attribute__((ext_vector_type(8)));

union Frag {
    v16bf    v;
    uint32_t u[8];
    uint4    q[2];
};

__device__ __forceinline__ uint16_t f2bf(float a) {
    uint32_t ua = __float_as_uint(a);
    return (uint16_t)((ua + 0x7FFFu + ((ua >> 16) & 1u)) >> 16);
}

// CDNA5 async global->LDS DMA (ASYNCcnt-tracked). LDS address = low 32 bits of
// the flat pointer.
__device__ __forceinline__ void async_b128(uint32_t lds, const void* gp) {
    asm volatile("global_load_async_to_lds_b128 %0, %1, off"
                 :: "v"(lds), "v"((uint64_t)(uintptr_t)gp)
                 : "memory");
}
__device__ __forceinline__ void wait_async0() {
    asm volatile("s_wait_asynccnt 0x0" ::: "memory");
}
__device__ __forceinline__ uint32_t lds_addr(const void* p) {
    return (uint32_t)(uintptr_t)p;
}

// ---------------------------------------------------------------------------
// dtype conversion kernels
// ---------------------------------------------------------------------------
__global__ void cvt_bf16(const float* __restrict__ src, uint16_t* __restrict__ dst, int n) {
    int i = blockIdx.x * blockDim.x + threadIdx.x;
    if (i < n) dst[i] = f2bf(src[i]);
}

// src: K x N row-major f32  ->  dst: N x K row-major bf16 (transposed)
__global__ void cvt_bf16_T(const float* __restrict__ src, uint16_t* __restrict__ dst,
                           int K, int N) {
    int i = blockIdx.x * blockDim.x + threadIdx.x;
    if (i < K * N) {
        int k = i / N, n = i % N;
        dst[(size_t)n * K + k] = f2bf(src[i]);
    }
}

// ---------------------------------------------------------------------------
// Wave-tiled bf16 GEMM:  out[M,N] = A[M,K] * B[K,N],  B given transposed (N,K).
// Block = 128 threads = 4 waves; wave w computes a 32x64 tile.
// Per k-step: 12 b128 loads feed 8 WMMAs (A-fragments reused across column tiles).
// ---------------------------------------------------------------------------
template <bool OUT_BF16>
__global__ __launch_bounds__(128) void gemm_bf16_wmma(
        const uint16_t* __restrict__ A,   // M x K row-major bf16
        const uint16_t* __restrict__ Bt,  // N x K row-major bf16
        void* __restrict__ out,
        int M, int N, int K) {
    const int lane = threadIdx.x & 31;
    const int w    = threadIdx.x >> 5;
    const int half = lane >> 4;
    const int ln   = lane & 15;
    const int m0   = blockIdx.x * 128 + w * 32;
    const int n0   = blockIdx.y * 64;

    v8f acc[2][4];
#pragma unroll
    for (int mt = 0; mt < 2; ++mt)
#pragma unroll
        for (int nb = 0; nb < 4; ++nb)
#pragma unroll
            for (int r = 0; r < 8; ++r) acc[mt][nb][r] = 0.0f;

    const uint16_t* Arow0 = A + (size_t)(m0 + ln) * K;
    const uint16_t* Arow1 = A + (size_t)(m0 + 16 + ln) * K;
    const uint16_t* Brow0 = Bt + (size_t)(n0 + ln) * K;

    for (int kc = 0; kc < K; kc += 32) {
        Frag af[2];
        af[0].q[0] = *(const uint4*)(Arow0 + kc + 8 * half);
        af[0].q[1] = *(const uint4*)(Arow0 + kc + 16 + 8 * half);
        af[1].q[0] = *(const uint4*)(Arow1 + kc + 8 * half);
        af[1].q[1] = *(const uint4*)(Arow1 + kc + 16 + 8 * half);
#pragma unroll
        for (int nb = 0; nb < 4; ++nb) {
            const uint16_t* Brow = Brow0 + (size_t)(nb * 16) * K;
            Frag bf;
            bf.q[0] = *(const uint4*)(Brow + kc + 8 * half);
            bf.q[1] = *(const uint4*)(Brow + kc + 16 + 8 * half);
            acc[0][nb] = __builtin_amdgcn_wmma_f32_16x16x32_bf16(
                false, af[0].v, false, bf.v, (short)0, acc[0][nb], false, false);
            acc[1][nb] = __builtin_amdgcn_wmma_f32_16x16x32_bf16(
                false, af[1].v, false, bf.v, (short)0, acc[1][nb], false, false);
        }
    }

#pragma unroll
    for (int mt = 0; mt < 2; ++mt) {
#pragma unroll
        for (int nb = 0; nb < 4; ++nb) {
            int n = n0 + nb * 16 + ln;
#pragma unroll
            for (int r = 0; r < 8; ++r) {
                int m = m0 + mt * 16 + r + 8 * half;
                float val = acc[mt][nb][r];
                if (OUT_BF16)
                    ((uint16_t*)out)[(size_t)m * N + n] = f2bf(val);
                else
                    ((float*)out)[(size_t)m * N + n] = val;
            }
        }
    }
}

// ---------------------------------------------------------------------------
// Fused causal multi-query flash attention (bf16 WMMA, f32 accumulate).
// grid = (T/128, H, B); block = 256 threads (8 waves, 16 queries each).
// KV interleaved: (B*T, 128) bf16 (K|V). Key blocks of 64, double-buffered
// via global_load_async_to_lds_b128 (ASYNCcnt); next block streams in during
// the WMMA/softmax phase of the current one. Row-sum reduction is deferred to
// the epilogue (only the max reduce is done online).
// ---------------------------------------------------------------------------
__global__ __launch_bounds__(256) void mqa_flash(
        const uint16_t* __restrict__ Qb,
        const uint16_t* __restrict__ KVb,
        uint16_t* __restrict__ Yb,
        int T) {
    __shared__ __align__(16) uint16_t ldsK[2][64 * 64];     // key-major (s, kd)
    __shared__ __align__(16) uint16_t ldsVraw[2][64 * 64];  // key-major (s, kd)
    __shared__ __align__(16) uint16_t ldsVt[64 * 64];       // transposed (kd, s)
    __shared__ __align__(16) uint16_t ldsP[8][16 * 64];

    const int tid  = threadIdx.x;
    const int w    = tid >> 5;
    const int lane = tid & 31;
    const int half = lane >> 4;
    const int ln   = lane & 15;
    const int b    = blockIdx.z;
    const int head = blockIdx.y;
    const int q0blk = blockIdx.x * 128;
    const int q0   = q0blk + w * 16;
    const size_t rowbase = (size_t)b * T;

    // scores scaled into log2 domain: S * (1/sqrt(64)) * log2(e)
    const float SCL = 0.125f * 1.4426950408889634f;

    // issue async K|V stage of a 64-key block: 4 x b128 per thread
    auto stage_async = [&](int s0, int buf) {
        const char* grow = (const char*)(KVb + (rowbase + s0) * 128);
        const uint32_t kbase = lds_addr(&ldsK[buf][0]);
        const uint32_t vbase = lds_addr(&ldsVraw[buf][0]);
#pragma unroll
        for (int t = 0; t < 2; ++t) {
            const int j = tid + t * 256;  // 512 x 16B chunks per matrix
            const int go = (j >> 3) * 256 + (j & 7) * 16;
            async_b128(kbase + j * 16, grow + go);
            async_b128(vbase + j * 16, grow + go + 128);
        }
    };

    // Q fragments: 16 queries x 64 dims, two K-chunks of 32
    Frag qf[2];
    {
        const uint16_t* Qrow = Qb + (rowbase + q0 + ln) * 1024 + head * 64;
#pragma unroll
        for (int c = 0; c < 2; ++c) {
            qf[c].q[0] = *(const uint4*)(Qrow + c * 32 + 8 * half);
            qf[c].q[1] = *(const uint4*)(Qrow + c * 32 + 16 + 8 * half);
        }
    }

    v8f o[4];
#pragma unroll
    for (int nb = 0; nb < 4; ++nb)
#pragma unroll
        for (int r = 0; r < 8; ++r) o[nb][r] = 0.0f;
    float rmax[8], rsum[8];  // rsum: per-lane partial (reduced in epilogue)
#pragma unroll
    for (int r = 0; r < 8; ++r) { rmax[r] = -1e30f; rsum[r] = 0.0f; }

    const int send = q0blk + 127;  // block-uniform causal bound
    stage_async(0, 0);

    for (int s0 = 0; s0 <= send; s0 += 64) {
        const int buf = (s0 >> 6) & 1;
        wait_async0();       // this wave's async loads for block s0 complete
        __syncthreads();     // whole block's loads done; prev compute finished

        if (s0 + 64 <= send) stage_async(s0 + 64, buf ^ 1);

        // transpose V (raw, key-major) -> ldsVt (dim-major); paired rows give
        // b32 stores instead of 2x b16
        {
            const uint32_t* vr = (const uint32_t*)&ldsVraw[buf][0];
            uint32_t* vt = (uint32_t*)ldsVt;
#pragma unroll
            for (int t = 0; t < 4; ++t) {
                const int p = tid + t * 256;  // 1024 (s-pair, kd-pair) tasks
                const int s2 = p >> 5, kd2 = p & 31;
                const uint32_t d0 = vr[(2 * s2) * 32 + kd2];
                const uint32_t d1 = vr[(2 * s2 + 1) * 32 + kd2];
                vt[kd2 * 64 + s2]      = (d0 & 0xFFFFu) | (d1 << 16);
                vt[kd2 * 64 + 32 + s2] = (d0 >> 16) | (d1 & 0xFFFF0000u);
            }
        }
        __syncthreads();

        // waves entirely below this key block skip compute (wave-uniform)
        if (s0 <= q0 + 15) {
            // ---- S = Q * K^T  (four 16x16 tiles over 64 keys) ----
            v8f sacc[4];
#pragma unroll
            for (int nb = 0; nb < 4; ++nb) {
#pragma unroll
                for (int r = 0; r < 8; ++r) sacc[nb][r] = 0.0f;
                const uint16_t* Krow = &ldsK[buf][0] + (nb * 16 + ln) * 64;
#pragma unroll
                for (int c = 0; c < 2; ++c) {
                    Frag kf;
                    kf.q[0] = *(const uint4*)(Krow + c * 32 + 8 * half);
                    kf.q[1] = *(const uint4*)(Krow + c * 32 + 16 + 8 * half);
                    sacc[nb] = __builtin_amdgcn_wmma_f32_16x16x32_bf16(
                        false, qf[c].v, false, kf.v, (short)0, sacc[nb], false, false);
                }
            }

            // ---- online softmax; masking only on the diagonal block ----
            const bool need_mask = (s0 + 63 > q0);
            uint16_t* P = ldsP[w];
#pragma unroll
            for (int r = 0; r < 8; ++r) {
                const int q = q0 + r + 8 * half;
                float vv[4];
#pragma unroll
                for (int nb = 0; nb < 4; ++nb) vv[nb] = sacc[nb][r] * SCL;
                if (need_mask) {
#pragma unroll
                    for (int nb = 0; nb < 4; ++nb)
                        if (s0 + nb * 16 + ln > q) vv[nb] = -1e30f;
                }
                float mloc = fmaxf(fmaxf(vv[0], vv[1]), fmaxf(vv[2], vv[3]));
#pragma unroll
                for (int off = 8; off >= 1; off >>= 1)
                    mloc = fmaxf(mloc, __shfl_xor(mloc, off, 32));
                const float newm  = fmaxf(rmax[r], mloc);
                // raw v_exp_f32: inputs are <= 0, underflow-to-zero is desired
                const float alpha = __builtin_amdgcn_exp2f(rmax[r] - newm);
                rmax[r] = newm;
                const int row = r + 8 * half;
                float ps = 0.0f;
#pragma unroll
                for (int nb = 0; nb < 4; ++nb) {
                    const float p = __builtin_amdgcn_exp2f(vv[nb] - newm);
                    ps += p;
                    P[row * 64 + nb * 16 + ln] = f2bf(p);
                }
                rsum[r] = rsum[r] * alpha + ps;  // per-lane partial sum
#pragma unroll
                for (int nb = 0; nb < 4; ++nb) o[nb][r] *= alpha;
            }

            // ---- P (16x64) as two A-fragments (same-wave LDS: in-order) ----
            Frag pf[2];
#pragma unroll
            for (int c = 0; c < 2; ++c) {
                pf[c].q[0] = *(const uint4*)(&P[ln * 64 + c * 32 + 8 * half]);
                pf[c].q[1] = *(const uint4*)(&P[ln * 64 + c * 32 + 16 + 8 * half]);
            }

            // ---- O += P * V  (8 WMMAs over 64 value dims x 64 keys) ----
#pragma unroll
            for (int nb = 0; nb < 4; ++nb) {
                const uint16_t* Vrow = ldsVt + (nb * 16 + ln) * 64;
#pragma unroll
                for (int c = 0; c < 2; ++c) {
                    Frag vf;
                    vf.q[0] = *(const uint4*)(Vrow + c * 32 + 8 * half);
                    vf.q[1] = *(const uint4*)(Vrow + c * 32 + 16 + 8 * half);
                    o[nb] = __builtin_amdgcn_wmma_f32_16x16x32_bf16(
                        false, pf[c].v, false, vf.v, (short)0, o[nb], false, false);
                }
            }
        }
    }

    // ---- reduce row sums once, normalize, emit Y (bf16) ----
    float inv[8];
#pragma unroll
    for (int r = 0; r < 8; ++r) {
        float s = rsum[r];
#pragma unroll
        for (int off = 8; off >= 1; off >>= 1)
            s += __shfl_xor(s, off, 32);
        inv[r] = __builtin_amdgcn_rcpf(s);
    }
#pragma unroll
    for (int nb = 0; nb < 4; ++nb) {
        const int n = nb * 16 + ln;
#pragma unroll
        for (int r = 0; r < 8; ++r) {
            const int t = q0 + r + 8 * half;
            const float val = o[nb][r] * inv[r];
            Yb[(rowbase + t) * 1024 + head * 64 + n] = f2bf(val);
        }
    }
}

// ---------------------------------------------------------------------------
// Launch
// ---------------------------------------------------------------------------
extern "C" void kernel_launch(void* const* d_in, const int* in_sizes, int n_in,
                              void* d_out, int out_size, void* d_ws, size_t ws_size,
                              hipStream_t stream) {
    (void)in_sizes; (void)n_in; (void)out_size; (void)ws_size;
    const float* x  = (const float*)d_in[0];
    const float* Wq = (const float*)d_in[1];
    const float* Wk = (const float*)d_in[2];
    const float* Wv = (const float*)d_in[3];
    const float* Wo = (const float*)d_in[4];
    float* out = (float*)d_out;

    const int B = 2, T = 2048, D = 1024, H = 16, KD = 64;
    const int M = B * T;  // 4096

    char* wsp = (char*)d_ws;
    size_t off = 0;
    auto alloc = [&](size_t bytes) -> void* {
        void* p = wsp + off;
        off = (off + bytes + 255) & ~(size_t)255;
        return p;
    };
    uint16_t* xb  = (uint16_t*)alloc((size_t)M * D * 2);
    uint16_t* Wqt = (uint16_t*)alloc((size_t)D * D * 2);
    uint16_t* KVt = (uint16_t*)alloc((size_t)(2 * KD) * D * 2);  // [K rows | V rows]
    uint16_t* Wot = (uint16_t*)alloc((size_t)D * D * 2);
    uint16_t* Qb  = (uint16_t*)alloc((size_t)M * D * 2);
    uint16_t* KVb = (uint16_t*)alloc((size_t)M * (2 * KD) * 2);  // per row: K|V
    uint16_t* Yb  = (uint16_t*)alloc((size_t)M * D * 2);

    // 1) precision conversion (+ weight transpose for contiguous B-fragments)
    cvt_bf16<<<(M * D + 255) / 256, 256, 0, stream>>>(x, xb, M * D);
    cvt_bf16_T<<<(D * D + 255) / 256, 256, 0, stream>>>(Wq, Wqt, D, D);
    cvt_bf16_T<<<(D * KD + 255) / 256, 256, 0, stream>>>(Wk, KVt, D, KD);
    cvt_bf16_T<<<(D * KD + 255) / 256, 256, 0, stream>>>(Wv, KVt + (size_t)KD * D, D, KD);
    cvt_bf16_T<<<(D * D + 255) / 256, 256, 0, stream>>>(Wo, Wot, D, D);

    // 2) projections (WMMA bf16, bf16 out); K and V fused into one GEMM
    gemm_bf16_wmma<true><<<dim3(M / 128, D / 64), 128, 0, stream>>>(xb, Wqt, Qb, M, D, D);
    gemm_bf16_wmma<true><<<dim3(M / 128, (2 * KD) / 64), 128, 0, stream>>>(
        xb, KVt, KVb, M, 2 * KD, D);

    // 3) fused causal multi-query flash attention (async double-buffered)
    mqa_flash<<<dim3(T / 128, H, B), 256, 0, stream>>>(Qb, KVb, Yb, T);

    // 4) output projection (WMMA bf16, f32 out)
    gemm_bf16_wmma<false><<<dim3(M / 128, D / 64), 128, 0, stream>>>(Yb, Wot, out, M, D, D);
}
